// GroupedQueryAttention_34583076667491
// MI455X (gfx1250) — matrix operand order repaired and checked by reading, not compiled, not run on hardware
//
#include <hip/hip_runtime.h>
#include <cstdint>
#include <cstddef>
#include <type_traits>

// Problem constants (match reference)
#define B_   2
#define S_   2048
#define H_   2048
#define NH_  32
#define NKV_ 8
#define HD_  64

typedef __attribute__((ext_vector_type(16))) __bf16 v16bf;
typedef __attribute__((ext_vector_type(8)))  __bf16 v8bf;
typedef __attribute__((ext_vector_type(4)))  __bf16 v4bf;
typedef __attribute__((ext_vector_type(8)))  float  v8f;
typedef __attribute__((ext_vector_type(4)))  unsigned int u32x4;
typedef __attribute__((ext_vector_type(8)))  int i32x8;
typedef __attribute__((ext_vector_type(4)))  int i32x4;

#if __has_builtin(__builtin_amdgcn_tensor_load_to_lds) && \
    __has_builtin(__builtin_amdgcn_s_wait_tensorcnt)
#define USE_TDM 1
#else
#define USE_TDM 0
#endif

__device__ __forceinline__ v16bf cat8(v8bf lo, v8bf hi) {
    return __builtin_shufflevector(lo, hi, 0, 1, 2, 3, 4, 5, 6, 7,
                                           8, 9, 10, 11, 12, 13, 14, 15);
}

#if USE_TDM
// Async 2D-tile DMA: global (row-major, stride_elems) -> LDS with per-row
// padding. bf16 elements (data_size=2B). Issued once per wave (EXEC ignored);
// caller gates to one wave and fences with s_wait_tensorcnt.
// pad_icode: row bytes = 4 * 2^(pad_icode+1); pad_acode: pad dwords - 1.
__device__ __forceinline__ void tdm_load_2d(
    const void* gsrc, unsigned lds_off, int tile_w, int tile_h,
    int stride_elems, unsigned pad_icode, unsigned pad_acode)
{
    uint64_t ga = (uint64_t)(uintptr_t)gsrc;
    u32x4 g0;
    g0[0] = 1u;                                        // count=1, user mode
    g0[1] = lds_off;                                   // LDS byte address
    g0[2] = (unsigned)ga;                              // global_addr[31:0]
    g0[3] = (unsigned)((ga >> 32) & 0x01FFFFFFu)       // global_addr[56:32]
          | (2u << 30);                                // type=2 ("image")
    i32x8 g1;
    g1[0] = (1 << 16)                                  // data_size = 2 bytes
          | (1 << 20)                                  // pad_enable
          | (int)(pad_icode << 22)                     // pad_interval
          | (int)(pad_acode << 25);                    // pad_amount
    g1[1] = (stride_elems & 0xFFFF) << 16;             // tensor_dim0[15:0]
    g1[2] = ((stride_elems >> 16) & 0xFFFF)            // tensor_dim0[31:16]
          | ((0x100000 & 0xFFFF) << 16);               // tensor_dim1[15:0] (big)
    g1[3] = (0x100000 >> 16)                           // tensor_dim1[31:16]
          | (tile_w << 16);                            // tile_dim0
    g1[4] = tile_h;                                    // tile_dim1 (tile_dim2=0)
    g1[5] = stride_elems;                              // tensor_dim0_stride[31:0]
    g1[6] = 0;
    g1[7] = 0;
    i32x4 z4 = {0, 0, 0, 0};
    i32x8 z8 = {0, 0, 0, 0, 0, 0, 0, 0};
    // amdgpu-toolchain (clang-23) 6-arg form: (g0, g1, g2, g3, extra, cpol)
    __builtin_amdgcn_tensor_load_to_lds(g0, g1, z4, z4, z8, 0);
}
#endif

// ---------------------------------------------------------------------------
// One-time fp32 -> bf16 conversion passes.
// ---------------------------------------------------------------------------
__global__ void cvt_bf16(const float* __restrict__ in, __bf16* __restrict__ out, int n)
{
    int i = (blockIdx.x * blockDim.x + threadIdx.x) * 4;
    if (i >= n) return;
    float4 v = *(const float4*)(in + i);
    v4bf o;
    o[0] = (__bf16)v.x; o[1] = (__bf16)v.y; o[2] = (__bf16)v.z; o[3] = (__bf16)v.w;
    *(v4bf*)(out + i) = o;
}

// W[K][N] fp32 -> WT[N][K] bf16 (transpose so GEMM B-fragments are K-contiguous)
__global__ void cvt_bf16_T(const float* __restrict__ W, __bf16* __restrict__ WT, int K, int N)
{
    int i = blockIdx.x * blockDim.x + threadIdx.x;
    if (i >= K * N) return;
    int n = i % N, k = i / N;                 // coalesced read of W
    WT[(size_t)n * K + k] = (__bf16)W[i];
}

// ---------------------------------------------------------------------------
// Tiled GEMM: C[M,N] = A[M,K] * B[K,N], A bf16 [M][K], B pre-transposed bf16
// [N][K]. 128x128 block, K-step 32, LDS double-buffered; tiles staged by the
// Tensor Data Mover (async, overlapped with WMMAs) when available.
// 8 waves; each wave: 64x32 = 4x2 accs of v_wmma_f32_16x16x32_bf16.
// ---------------------------------------------------------------------------
#define BM 128
#define BN 128
#define BK 32

template <bool OUT_BF16>
__global__ __launch_bounds__(256) void gemm_bf16(
    const __bf16* __restrict__ A, const __bf16* __restrict__ Bt,
    void* __restrict__ Cptr, int M, int N, int K)
{
    __shared__ __bf16 As[2][BM][BK + 8];    // row = 80B (16B aligned), TDM pad 8
    __shared__ __bf16 BsT[2][BN][BK + 8];

    const int tid  = threadIdx.x;
    const int lane = tid & 31;
    const int swave = __builtin_amdgcn_readfirstlane(tid) >> 5;   // scalar wave id
    const int bm   = blockIdx.y * BM;
    const int bn   = blockIdx.x * BN;
    const int wm   = (swave >> 2) * 64;
    const int wn   = (swave & 3) * 32;

    const int arow = lane & 15;
    const int akh  = (lane >> 4) * 8;
    const int bcol = lane & 15;
    const int bkh  = (lane >> 4) * 8;
    const int crow = (lane >> 4) * 8;

    auto stage = [&](int kt, int buf) {
        const int k0 = kt * BK;
#if USE_TDM
        if (swave == 0) {   // scalar branch; TDM ignores EXEC
            tdm_load_2d(A  + (size_t)bm * K + k0,
                        (unsigned)(uintptr_t)&As[buf][0][0], BK, BM, K, 3, 3);
            tdm_load_2d(Bt + (size_t)bn * K + k0,
                        (unsigned)(uintptr_t)&BsT[buf][0][0], BK, BN, K, 3, 3);
        }
#else
        const int sr = tid >> 1;
        const int sh = (tid & 1) * 16;
        const v8bf* sa = (const v8bf*)(A  + (size_t)(bm + sr) * K + k0 + sh);
        v8bf a0 = sa[0], a1 = sa[1];
        const v8bf* sb = (const v8bf*)(Bt + (size_t)(bn + sr) * K + k0 + sh);
        v8bf b0 = sb[0], b1 = sb[1];
        *(v8bf*)&As[buf][sr][sh]      = a0;
        *(v8bf*)&As[buf][sr][sh + 8]  = a1;
        *(v8bf*)&BsT[buf][sr][sh]     = b0;
        *(v8bf*)&BsT[buf][sr][sh + 8] = b1;
#endif
    };
    auto stage_fence = [&]() {
#if USE_TDM
        if (swave == 0) __builtin_amdgcn_s_wait_tensorcnt(0);
#endif
    };

    v8f acc[4][2] = {};
    const int nt = K / BK;

    stage(0, 0);
    stage_fence();
    __syncthreads();

    for (int kt = 0; kt < nt; ++kt) {
        const int cur = kt & 1;
        if (kt + 1 < nt) stage(kt + 1, cur ^ 1);   // async; overlaps WMMAs

        v16bf a[4], b[2];
        #pragma unroll
        for (int t = 0; t < 4; ++t) {
            v8bf lo = *(const v8bf*)&As[cur][wm + t * 16 + arow][akh];
            v8bf hi = *(const v8bf*)&As[cur][wm + t * 16 + arow][16 + akh];
            a[t] = cat8(lo, hi);
        }
        #pragma unroll
        for (int t = 0; t < 2; ++t) {
            v8bf lo = *(const v8bf*)&BsT[cur][wn + t * 16 + bcol][bkh];
            v8bf hi = *(const v8bf*)&BsT[cur][wn + t * 16 + bcol][bkh + 16];
            b[t] = cat8(lo, hi);
        }
        #pragma unroll
        for (int i = 0; i < 4; ++i)
            #pragma unroll
            for (int j = 0; j < 2; ++j)
                acc[i][j] = __builtin_amdgcn_wmma_f32_16x16x32_bf16(
                    false, a[i], false, b[j], (short)0, acc[i][j], false, false);

        stage_fence();           // next buffer ready before the barrier
        __syncthreads();
    }

    #pragma unroll
    for (int i = 0; i < 4; ++i)
        #pragma unroll
        for (int j = 0; j < 2; ++j)
            #pragma unroll
            for (int r = 0; r < 8; ++r) {
                size_t off = (size_t)(bm + wm + i * 16 + crow + r) * N
                           + (bn + wn + j * 16 + bcol);
                if (OUT_BF16) ((__bf16*)Cptr)[off] = (__bf16)acc[i][j][r];
                else          ((float*)Cptr)[off]  = acc[i][j][r];
            }
}

// ---------------------------------------------------------------------------
// RoPE on fp32 q/k from the projection GEMMs; emits bf16 q/k for attention.
// ---------------------------------------------------------------------------
__global__ void rope_cvt_kernel(const float* __restrict__ qf, const float* __restrict__ kf,
                                __bf16* __restrict__ qb, __bf16* __restrict__ kb)
{
    const int halfd = HD_ / 2;             // 32
    const int HTOT  = NH_ + NKV_;          // 40
    int idx = blockIdx.x * blockDim.x + threadIdx.x;
    int total = B_ * S_ * HTOT * halfd;
    if (idx >= total) return;

    int d    = idx % halfd;
    int rest = idx / halfd;
    int h    = rest % HTOT;
    int bs   = rest / HTOT;
    int s    = bs % S_;

    float inv_freq = __expf(-((float)(2 * d) / (float)HD_) * __logf(10000.0f));
    float ang = (float)s * inv_freq;
    float sn, cs;
    __sincosf(ang, &sn, &cs);

    size_t off = (h < NH_) ? ((size_t)bs * NH_ + h) * HD_
                           : ((size_t)bs * NKV_ + (h - NH_)) * HD_;
    const float* src = (h < NH_) ? (qf + off) : (kf + off);
    __bf16*      dst = (h < NH_) ? (qb + off) : (kb + off);
    float x1 = src[d];
    float x2 = src[d + halfd];
    dst[d]         = (__bf16)(x1 * cs - x2 * sn);
    dst[d + halfd] = (__bf16)(x2 * cs + x1 * sn);
}

// ---------------------------------------------------------------------------
// Flash attention (causal + pad mask), GQA. bf16 in/out. 8 waves, wave owns
// 16 queries; 32-key blocks. K tile DMA'd by TDM (row-major); V staged
// transposed by the block. Causal masking only on the (scalar-branched)
// diagonal blocks; pad mask folded into an additive bias. P repack via
// per-wave LDS + s_wait_dscnt.
// ---------------------------------------------------------------------------
__global__ __launch_bounds__(256) void attn_fwd(
    const __bf16* __restrict__ Qg, const __bf16* __restrict__ Kg,
    const __bf16* __restrict__ Vg, const int* __restrict__ am,
    __bf16* __restrict__ Og)
{
    __shared__ __bf16 Kb[32][HD_ + 8];     // [key][dim], TDM row pad 8
    __shared__ __bf16 VbT[HD_][32 + 8];    // [dim][key]
    __shared__ __bf16 Pb[8][16][40];       // per-wave P repack scratch

    const int tid  = threadIdx.x;
    const int wave = tid >> 5;
    const int lane = tid & 31;
    const int swave = __builtin_amdgcn_readfirstlane(tid) >> 5;
    const int b    = blockIdx.z;
    const int h    = blockIdx.y;
    const int kvh  = h >> 2;                       // NH/NKV = 4
    const int qt   = blockIdx.x * 128 + wave * 16;
    const int qts  = blockIdx.x * 128 + swave * 16;   // scalar copy of qt

    const int arow = lane & 15;
    const int akh  = (lane >> 4) * 8;
    const int bcol = lane & 15;
    const int bkh  = (lane >> 4) * 8;
    const int crow = (lane >> 4) * 8;

    // Q fragments: direct bf16 vector loads (A-layout = 2x 16B runs per frag).
    const __bf16* qrow = Qg + ((size_t)(b * S_ + qt + arow) * (NH_ * HD_)) + h * HD_;
    v16bf qa[2];
    #pragma unroll
    for (int t = 0; t < 2; ++t) {
        v8bf lo = *(const v8bf*)(qrow + t * 32 + akh);
        v8bf hi = *(const v8bf*)(qrow + t * 32 + 16 + akh);
        qa[t] = cat8(lo, hi);
    }

    v8f acc[4] = {};
    float mrow[8], lrow[8];
    #pragma unroll
    for (int r = 0; r < 8; ++r) { mrow[r] = -3.0e38f; lrow[r] = 0.0f; }

    const int skr = tid >> 3, skc = (tid & 7) * 8;   // K fallback staging
    const int svd = tid & 63, svr = (tid >> 6) * 8;  // V transpose staging

    const int nkb = blockIdx.x * 4 + 4;
    for (int kb = 0; kb < nkb; ++kb) {
        const int kbase = kb * 32;
        // ---- K tile: async TDM (rows 128B + 16B pad) ----
#if USE_TDM
        if (swave == 0)
            tdm_load_2d(Kg + ((size_t)(b * S_ + kbase) * (NKV_ * HD_)) + kvh * HD_,
                        (unsigned)(uintptr_t)&Kb[0][0], HD_, 32, NKV_ * HD_, 4, 3);
#else
        *(v8bf*)&Kb[skr][skc] = *(const v8bf*)(
            Kg + ((size_t)(b * S_ + kbase + skr) * (NKV_ * HD_)) + kvh * HD_ + skc);
#endif
        // ---- V tile transposed (manual: TDM has no transpose) ----
        {
            v8bf tmp;
            #pragma unroll
            for (int j = 0; j < 8; ++j)
                tmp[j] = Vg[((size_t)(b * S_ + kbase + svr + j) * (NKV_ * HD_)) + kvh * HD_ + svd];
            *(v8bf*)&VbT[svd][svr] = tmp;
        }
        if (kb + 1 < nkb)   // emits global_prefetch_b8
            __builtin_prefetch(
                Vg + ((size_t)(b * S_ + kbase + 32 + skr) * (NKV_ * HD_)) + kvh * HD_ + skc, 0, 1);
#if USE_TDM
        if (swave == 0) __builtin_amdgcn_s_wait_tensorcnt(0);
#endif
        __syncthreads();

        if (kbase <= qts + 15) {           // scalar causal skip
            // ---- S = Q * K^T : two 16x16 score tiles over 32 keys ----
            v8f sc[2] = {};
            #pragma unroll
            for (int n = 0; n < 2; ++n) {
                #pragma unroll
                for (int kk = 0; kk < 2; ++kk) {
                    v8bf lo = *(const v8bf*)&Kb[n * 16 + bcol][kk * 32 + bkh];
                    v8bf hi = *(const v8bf*)&Kb[n * 16 + bcol][kk * 32 + bkh + 16];
                    sc[n] = __builtin_amdgcn_wmma_f32_16x16x32_bf16(
                        false, qa[kk], false, cat8(lo, hi), (short)0, sc[n], false, false);
                }
            }
            // ---- masks + online softmax ----
            const int key0 = kbase + bcol;
            const int key1 = kbase + 16 + bcol;
            const float pb0 = (am[b * S_ + key0] == 0) ? -3.0e38f : 0.0f;
            const float pb1 = (am[b * S_ + key1] == 0) ? -3.0e38f : 0.0f;

            auto softmax_update = [&](auto CAUSAL) {
                #pragma unroll
                for (int r = 0; r < 8; ++r) {
                    float s0 = fmaf(sc[0][r], 0.125f, pb0);   // 1/sqrt(64) + pad
                    float s1 = fmaf(sc[1][r], 0.125f, pb1);
                    if constexpr (decltype(CAUSAL)::value) {  // diagonal only
                        int qidx = qt + crow + r;
                        if (key0 > qidx) s0 = -3.0e38f;
                        if (key1 > qidx) s1 = -3.0e38f;
                    }
                    float mx = fmaxf(s0, s1);
                    #pragma unroll
                    for (int d = 1; d < 16; d <<= 1) mx = fmaxf(mx, __shfl_xor(mx, d, 32));
                    float mnew = fmaxf(mrow[r], mx);
                    float p0 = __expf(s0 - mnew);
                    float p1 = __expf(s1 - mnew);
                    float rs = p0 + p1;
                    #pragma unroll
                    for (int d = 1; d < 16; d <<= 1) rs += __shfl_xor(rs, d, 32);
                    float alpha = __expf(mrow[r] - mnew);
                    lrow[r] = lrow[r] * alpha + rs;
                    mrow[r] = mnew;
                    #pragma unroll
                    for (int t = 0; t < 4; ++t) acc[t][r] = acc[t][r] * alpha;
                    Pb[wave][crow + r][bcol]      = (__bf16)p0;
                    Pb[wave][crow + r][16 + bcol] = (__bf16)p1;
                }
            };
            if (kbase + 31 > qts)                       // scalar branch: diagonal
                softmax_update(std::integral_constant<bool, true>{});
            else                                        // interior: compare-free
                softmax_update(std::integral_constant<bool, false>{});

            asm volatile("s_wait_dscnt 0" ::: "memory");   // order Pb write -> read
            v8bf plo = *(const v8bf*)&Pb[wave][arow][akh];
            v8bf phi = *(const v8bf*)&Pb[wave][arow][16 + akh];
            v16bf pa = cat8(plo, phi);
            // ---- O += P * V ----
            #pragma unroll
            for (int t = 0; t < 4; ++t) {
                v8bf lo = *(const v8bf*)&VbT[t * 16 + bcol][bkh];
                v8bf hi = *(const v8bf*)&VbT[t * 16 + bcol][bkh + 16];
                acc[t] = __builtin_amdgcn_wmma_f32_16x16x32_bf16(
                    false, pa, false, cat8(lo, hi), (short)0, acc[t], false, false);
            }
        }
        __syncthreads();
    }

    // Normalize; store bf16 O [B,S,NH,HD] for the output-projection GEMM.
    #pragma unroll
    for (int r = 0; r < 8; ++r) {
        float inv = (lrow[r] > 0.0f) ? (1.0f / lrow[r]) : 0.0f;
        int qidx = qt + crow + r;
        #pragma unroll
        for (int t = 0; t < 4; ++t)
            Og[((size_t)(b * S_ + qidx) * (NH_ * HD_)) + h * HD_ + t * 16 + bcol] =
                (__bf16)(acc[t][r] * inv);
    }
}

// ---------------------------------------------------------------------------
// Orchestration.
// ---------------------------------------------------------------------------
extern "C" void kernel_launch(void* const* d_in, const int* in_sizes, int n_in,
                              void* d_out, int out_size, void* d_ws, size_t ws_size,
                              hipStream_t stream)
{
    const float* hs    = (const float*)d_in[0];
    const int*   amask = (const int*)d_in[1];
    const float* Wq    = (const float*)d_in[2];
    const float* Wk    = (const float*)d_in[3];
    const float* Wv    = (const float*)d_in[4];
    const float* Wo    = (const float*)d_in[5];

    const int M  = B_ * S_;                       // 4096
    const int Nq = NH_ * HD_;                     // 2048
    const int Nk = NKV_ * HD_;                    // 512

    __bf16* hsb = (__bf16*)d_ws;                  // [M][H]
    __bf16* wqT = hsb + (size_t)M * H_;           // [Nq][H]
    __bf16* wkT = wqT + (size_t)Nq * H_;          // [Nk][H]
    __bf16* wvT = wkT + (size_t)Nk * H_;          // [Nk][H]
    __bf16* woT = wvT + (size_t)Nk * H_;          // [H][Nq]
    float*  qf  = (float*)(woT + (size_t)H_ * Nq);
    float*  kf  = qf + (size_t)M * Nq;
    __bf16* qb  = (__bf16*)(kf + (size_t)M * Nk);
    __bf16* kb  = qb + (size_t)M * Nq;
    __bf16* vb  = kb + (size_t)M * Nk;
    __bf16* ob  = vb + (size_t)M * Nk;

    dim3 blk(256);

    {
        int n = M * H_;
        cvt_bf16<<<(n / 4 + 255) / 256, blk, 0, stream>>>(hs, hsb, n);
        cvt_bf16_T<<<(H_ * Nq + 255) / 256, blk, 0, stream>>>(Wq, wqT, H_, Nq);
        cvt_bf16_T<<<(H_ * Nk + 255) / 256, blk, 0, stream>>>(Wk, wkT, H_, Nk);
        cvt_bf16_T<<<(H_ * Nk + 255) / 256, blk, 0, stream>>>(Wv, wvT, H_, Nk);
        cvt_bf16_T<<<(Nq * H_ + 255) / 256, blk, 0, stream>>>(Wo, woT, Nq, H_);
    }

    gemm_bf16<false><<<dim3(Nq / BN, M / BM), blk, 0, stream>>>(hsb, wqT, qf, M, Nq, H_);
    gemm_bf16<false><<<dim3(Nk / BN, M / BM), blk, 0, stream>>>(hsb, wkT, kf, M, Nk, H_);
    gemm_bf16<true ><<<dim3(Nk / BN, M / BM), blk, 0, stream>>>(hsb, wvT, vb, M, Nk, H_);

    int rope_total = B_ * S_ * (NH_ + NKV_) * (HD_ / 2);
    rope_cvt_kernel<<<(rope_total + 255) / 256, blk, 0, stream>>>(qf, kf, qb, kb);

    attn_fwd<<<dim3(S_ / 128, NH_, B_), blk, 0, stream>>>(qb, kb, vb, amask, ob);

    gemm_bf16<false><<<dim3(H_ / BN, M / BM), blk, 0, stream>>>(ob, woT, (float*)d_out, M, H_, H_);
}
